// Attention_63024350101937
// MI455X (gfx1250) — matrix operand order repaired and checked by reading, not compiled, not run on hardware
//
#include <hip/hip_runtime.h>
#include <hip/hip_bf16.h>

#define EMB   1024
#define NSEQ  2048
#define HEADS 16
#define HD    64
#define BATCH 2
#define SOFTMAX_SCALE 0.125f   // 1/sqrt(64)

typedef __attribute__((ext_vector_type(16))) _Float16 v16h;
typedef __attribute__((ext_vector_type(8)))  _Float16 v8h;
typedef __attribute__((ext_vector_type(4)))  _Float16 v4h;
typedef __attribute__((ext_vector_type(8)))  float    v8f;

// ---------------------------------------------------------------------------
// WMMA helpers (CDNA5 16x16x32 f16 -> f32 accum)
// ---------------------------------------------------------------------------
static __device__ __forceinline__ v8f wmma16(v16h a, v16h b, v8f c) {
  return __builtin_amdgcn_wmma_f32_16x16x32_f16(false, a, false, b, (short)0, c,
                                                false, false);
}

// A-matrix 16x32 f16 fragment from row-major source.
// Layout used here (contiguous k, matched by load_b16): lane L holds row M=L%16; lanes<16: K {0..15}; lanes>=16:
// K {16..31}. Two 16B contiguous chunks per lane.
static __device__ __forceinline__ v16h load_a16(const _Float16* base, int ld,
                                                int lane) {
  const _Float16* p = base + (size_t)(lane & 15) * ld + ((lane >> 4) << 4);
  v8h lo = *(const v8h*)p;
  v8h hi = *(const v8h*)(p + 8);
  v16h a;
#pragma unroll
  for (int i = 0; i < 8; ++i) { a[i] = lo[i]; a[i + 8] = hi[i]; }
  return a;
}

// B-matrix 32x16 f16 fragment. Layout used here (contiguous k, matched by load_a16): lane holds column N=L%16,
// K = (L<16 ? 0 : 16) + [0..15] contiguous. Source stored N-major with
// leading dim `ld` over K (i.e. element (k,n) at base[n*ld + k]):
// 32B contiguous per lane.
static __device__ __forceinline__ v16h load_b16(const _Float16* base, int ld,
                                                int lane) {
  const _Float16* p = base + (size_t)(lane & 15) * ld + ((lane >> 4) << 4);
  return *(const v16h*)p;
}

// ---------------------------------------------------------------------------
// f32 -> f16 conversion kernels
// ---------------------------------------------------------------------------
__global__ void cvt_f16_kernel(const float* __restrict__ in,
                               _Float16* __restrict__ out, int n4) {
  int i = blockIdx.x * blockDim.x + threadIdx.x;
  if (i >= n4) return;
  float4 v = ((const float4*)in)[i];
  v4h h;
  h[0] = (_Float16)v.x; h[1] = (_Float16)v.y;
  h[2] = (_Float16)v.z; h[3] = (_Float16)v.w;
  ((v4h*)out)[i] = h;
}

// W [K=1024][N=1024] f32 row-major  ->  Wt [N][K] f16 (so GEMM B-frags are
// contiguous per lane).
__global__ void cvt_wt_kernel(const float* __restrict__ W,
                              _Float16* __restrict__ Wt) {
  int idx = blockIdx.x * blockDim.x + threadIdx.x;
  int k = idx >> 10, n = idx & 1023;
  Wt[(size_t)n * EMB + k] = (_Float16)W[(size_t)k * EMB + n];
}

// ---------------------------------------------------------------------------
// GEMM: C[M,1024] = A[M,1024](f16) @ W(Wt transposed, f16) + bias(f32)
// mode 0: out f16 row-major   (Q/K projections)
// mode 1: out f16 transposed per batch: out[b][col][n]  (V projection)
// mode 2: out f32 row-major   (final projection -> d_out)
// Block = 8 waves; block tile 128(M) x 128(N); wave tile 16 x 128.
// sched_barrier(0) pins the shape: all A/B fragment loads issue as one cluster before
// the 8 WMMAs, so LOADcnt waits are staggered and the next iteration's loads
// are free to hoist above the WMMA tail (software pipelining).
// ---------------------------------------------------------------------------
__global__ __launch_bounds__(256) void gemm_kernel(
    const _Float16* __restrict__ A, const _Float16* __restrict__ Wt,
    const float* __restrict__ bias, _Float16* __restrict__ outH,
    float* __restrict__ outF, int mode) {
  const int wave = threadIdx.x >> 5;
  const int lane = threadIdx.x & 31;
  const int m0 = blockIdx.x * 128 + wave * 16;
  const int n0 = blockIdx.y * 128;

  v8f acc[8] = {};
  const _Float16* Arow = A + (size_t)m0 * EMB;
  for (int k0 = 0; k0 < EMB; k0 += 32) {
    v16h a = load_a16(Arow + k0, EMB, lane);
    if (k0 + 32 < EMB)  // prefetch next A tile -> global_prefetch_b8
      __builtin_prefetch(Arow + (size_t)(lane & 15) * EMB + k0 + 32, 0, 1);
    v16h bf[8];
#pragma unroll
    for (int t = 0; t < 8; ++t)
      bf[t] = load_b16(Wt + (size_t)(n0 + t * 16) * EMB + k0, EMB, lane);
    // Fence: nothing crosses -> loads cluster ahead of the WMMA block.
    __builtin_amdgcn_sched_barrier(0);
#pragma unroll
    for (int t = 0; t < 8; ++t)
      acc[t] = wmma16(a, bf[t], acc[t]);
  }

  const int rowoff = (lane >> 4) << 3;  // C layout: lanes>=16 hold rows 8..15
  const int col = lane & 15;
#pragma unroll
  for (int t = 0; t < 8; ++t) {
    const float bv = bias[n0 + t * 16 + col];
#pragma unroll
    for (int r = 0; r < 8; ++r) {
      const float v = acc[t][r] + bv;
      const int row = m0 + r + rowoff;
      const int cc = n0 + t * 16 + col;
      if (mode == 0) {
        outH[(size_t)row * EMB + cc] = (_Float16)v;
      } else if (mode == 1) {
        const int bb = row >> 11;            // row / 2048
        const int nn = row & (NSEQ - 1);     // row % 2048
        outH[((size_t)bb * EMB + cc) * NSEQ + nn] = (_Float16)v;
      } else {
        outF[(size_t)row * EMB + cc] = v;
      }
    }
  }
}

// ---------------------------------------------------------------------------
// Flash attention. Grid: (B*H, NSEQ/128). 8 waves/block, wave = 16 q rows.
// Qh/Kh: [B,N,EMB] f16 row-major. Vt: [B,EMB,N] f16 (transposed).
// Oh: [B,N,EMB] f16. mbuf/lbuf: per-row softmax stats for batch 0.
// K and V fragment loads are fenced ahead of the QK^T WMMAs; the V loads stay
// in flight underneath the softmax VALU chain + LDS round-trip.
// ---------------------------------------------------------------------------
__global__ __launch_bounds__(256) void flash_attn_kernel(
    const _Float16* __restrict__ Qh, const _Float16* __restrict__ Kh,
    const _Float16* __restrict__ Vt, _Float16* __restrict__ Oh,
    float* __restrict__ mbuf, float* __restrict__ lbuf) {
  __shared__ _Float16 plds[8][16][40];  // per-wave 16x32 P tile, padded pitch

  const int wave = threadIdx.x >> 5;
  const int lane = threadIdx.x & 31;
  const int b = blockIdx.x >> 4;
  const int h = blockIdx.x & 15;
  const int q0 = blockIdx.y * 128 + wave * 16;
  const int rowoff = (lane >> 4) << 3;
  const int col = lane & 15;

  const _Float16* Qbase = Qh + ((size_t)(b * NSEQ + q0) * EMB) + h * HD;
  const v16h aq0 = load_a16(Qbase, EMB, lane);       // d 0..31
  const v16h aq1 = load_a16(Qbase + 32, EMB, lane);  // d 32..63

  float m[8], l[8], alpha[8];
  v8f o[4] = {};
#pragma unroll
  for (int r = 0; r < 8; ++r) { m[r] = -3.0e38f; l[r] = 0.0f; }

  const _Float16* Kb = Kh + (size_t)b * NSEQ * EMB + h * HD;
  const _Float16* Vb = Vt + ((size_t)(b * EMB + h * HD)) * NSEQ;

  for (int kt = 0; kt < NSEQ / 32; ++kt) {
    const _Float16* K0 = Kb + (size_t)(kt * 32) * EMB;
    const _Float16* V0 = Vb + kt * 32;

    // Issue all K + V fragment loads as one cluster.
    v16h kb0 = load_b16(K0, EMB, lane);
    v16h kb1 = load_b16(K0 + 32, EMB, lane);
    v16h kb2 = load_b16(K0 + 16 * EMB, EMB, lane);
    v16h kb3 = load_b16(K0 + 16 * EMB + 32, EMB, lane);
    v16h vb[4];
#pragma unroll
    for (int j = 0; j < 4; ++j)
      vb[j] = load_b16(V0 + (size_t)(j * 16) * NSEQ, NSEQ, lane);
    __builtin_amdgcn_sched_barrier(0);

    // S = Q K^T for 32 keys (two 16x16 tiles), D=64 => 2 WMMA per tile
    v8f s0 = {}, s1 = {};
    s0 = wmma16(aq0, kb0, s0);
    s0 = wmma16(aq1, kb1, s0);
    s1 = wmma16(aq0, kb2, s1);
    s1 = wmma16(aq1, kb3, s1);

    // online softmax over the 32 new keys
#pragma unroll
    for (int r = 0; r < 8; ++r) {
      float t0 = s0[r] * SOFTMAX_SCALE;
      float t1 = s1[r] * SOFTMAX_SCALE;
      float mx = fmaxf(t0, t1);
#pragma unroll
      for (int d = 1; d < 16; d <<= 1) mx = fmaxf(mx, __shfl_xor(mx, d));
      float mn = fmaxf(m[r], mx);
      float a = __expf(m[r] - mn);
      float p0 = __expf(t0 - mn);
      float p1 = __expf(t1 - mn);
      float sum = p0 + p1;
#pragma unroll
      for (int d = 1; d < 16; d <<= 1) sum += __shfl_xor(sum, d);
      l[r] = l[r] * a + sum;
      m[r] = mn;
      alpha[r] = a;
      // C-layout -> row-major P tile in this wave's private LDS region
      plds[wave][r + rowoff][col] = (_Float16)p0;
      plds[wave][r + rowoff][col + 16] = (_Float16)p1;
    }
#pragma unroll
    for (int j = 0; j < 4; ++j)
#pragma unroll
      for (int r = 0; r < 8; ++r) o[j][r] *= alpha[r];

    // In-wave LDS ordering: drain DS counter before re-reading as A-frag.
    asm volatile("s_wait_dscnt 0" ::: "memory");

    v16h ap = load_a16(&plds[wave][0][0], 40, lane);  // P as 16x32 A matrix
#pragma unroll
    for (int j = 0; j < 4; ++j)  // O[16q x 64d] += P @ V
      o[j] = wmma16(ap, vb[j], o[j]);
  }

  float il[8];
#pragma unroll
  for (int r = 0; r < 8; ++r) il[r] = 1.0f / l[r];
  _Float16* Ob = Oh + ((size_t)(b * NSEQ + q0) * EMB) + h * HD;
#pragma unroll
  for (int j = 0; j < 4; ++j)
#pragma unroll
    for (int r = 0; r < 8; ++r)
      Ob[(size_t)(r + rowoff) * EMB + j * 16 + col] = (_Float16)(o[j][r] * il[r]);

  if (b == 0 && col == 0) {  // lanes 0 (rows 0-7) and 16 (rows 8-15)
#pragma unroll
    for (int r = 0; r < 8; ++r) {
      int row = q0 + r + rowoff;
      mbuf[h * NSEQ + row] = m[r];
      lbuf[h * NSEQ + row] = l[r];
    }
  }
}

// ---------------------------------------------------------------------------
// total_attention_weights[q,k] = sum_h exp(scale*QK^T - m)/l  (batch 0).
// Grid: (NSEQ/128, NSEQ/128); wave tile = 16q x 128k, accumulated over heads.
// ---------------------------------------------------------------------------
__global__ __launch_bounds__(256) void attn_weights_kernel(
    const _Float16* __restrict__ Qh, const _Float16* __restrict__ Kh,
    const float* __restrict__ mbuf, const float* __restrict__ lbuf,
    float* __restrict__ outw) {
  const int wave = threadIdx.x >> 5;
  const int lane = threadIdx.x & 31;
  const int q0 = blockIdx.x * 128 + wave * 16;
  const int k0 = blockIdx.y * 128;
  const int rowoff = (lane >> 4) << 3;
  const int col = lane & 15;

  v8f acc[8] = {};
  for (int h = 0; h < HEADS; ++h) {
    const _Float16* Qbase = Qh + (size_t)q0 * EMB + h * HD;  // batch 0
    v16h aq0 = load_a16(Qbase, EMB, lane);
    v16h aq1 = load_a16(Qbase + 32, EMB, lane);
    float m8[8], il8[8];
#pragma unroll
    for (int r = 0; r < 8; ++r) {
      int row = q0 + r + rowoff;
      m8[r] = mbuf[h * NSEQ + row];
      il8[r] = 1.0f / lbuf[h * NSEQ + row];
    }
#pragma unroll 2
    for (int t = 0; t < 8; ++t) {
      const _Float16* Kbase = Kh + (size_t)(k0 + t * 16) * EMB + h * HD;
      v16h kb0 = load_b16(Kbase, EMB, lane);
      v16h kb1 = load_b16(Kbase + 32, EMB, lane);
      __builtin_amdgcn_sched_barrier(0);
      v8f s = {};
      s = wmma16(aq0, kb0, s);
      s = wmma16(aq1, kb1, s);
#pragma unroll
      for (int r = 0; r < 8; ++r)
        acc[t][r] += __expf(s[r] * SOFTMAX_SCALE - m8[r]) * il8[r];
    }
  }
#pragma unroll
  for (int t = 0; t < 8; ++t)
#pragma unroll
    for (int r = 0; r < 8; ++r)
      outw[(size_t)(q0 + r + rowoff) * NSEQ + (k0 + t * 16 + col)] = acc[t][r];
}

// ---------------------------------------------------------------------------
// Host launcher
// ---------------------------------------------------------------------------
extern "C" void kernel_launch(void* const* d_in, const int* in_sizes, int n_in,
                              void* d_out, int out_size, void* d_ws,
                              size_t ws_size, hipStream_t stream) {
  (void)in_sizes; (void)n_in; (void)out_size; (void)ws_size;
  const float* q  = (const float*)d_in[0];
  const float* k  = (const float*)d_in[1];
  const float* v  = (const float*)d_in[2];
  const float* Wq = (const float*)d_in[3];
  const float* bq = (const float*)d_in[4];
  const float* Wk = (const float*)d_in[5];
  const float* bk = (const float*)d_in[6];
  const float* Wv = (const float*)d_in[7];
  const float* bv = (const float*)d_in[8];
  const float* Wo = (const float*)d_in[9];
  const float* bo = (const float*)d_in[10];

  char* ws = (char*)d_ws;
  const size_t SZ_ACT = (size_t)BATCH * NSEQ * EMB * sizeof(_Float16);  // 8 MB
  const size_t SZ_W   = (size_t)EMB * EMB * sizeof(_Float16);           // 2 MB
  _Float16* q16 = (_Float16*)(ws);
  _Float16* k16 = (_Float16*)(ws + SZ_ACT);
  _Float16* v16 = (_Float16*)(ws + 2 * SZ_ACT);
  _Float16* Qh  = (_Float16*)(ws + 3 * SZ_ACT);
  _Float16* Kh  = (_Float16*)(ws + 4 * SZ_ACT);
  _Float16* Vt  = (_Float16*)(ws + 5 * SZ_ACT);
  _Float16* Oh  = (_Float16*)(ws + 6 * SZ_ACT);
  _Float16* Wqt = (_Float16*)(ws + 7 * SZ_ACT);
  _Float16* Wkt = (_Float16*)(ws + 7 * SZ_ACT + SZ_W);
  _Float16* Wvt = (_Float16*)(ws + 7 * SZ_ACT + 2 * SZ_W);
  _Float16* Wot = (_Float16*)(ws + 7 * SZ_ACT + 3 * SZ_W);
  float* mbuf   = (float*)(ws + 7 * SZ_ACT + 4 * SZ_W);
  float* lbuf   = mbuf + HEADS * NSEQ;

  const int n4 = BATCH * NSEQ * EMB / 4;  // 1,048,576 float4 per tensor
  cvt_f16_kernel<<<n4 / 256, 256, 0, stream>>>(q, q16, n4);
  cvt_f16_kernel<<<n4 / 256, 256, 0, stream>>>(k, k16, n4);
  cvt_f16_kernel<<<n4 / 256, 256, 0, stream>>>(v, v16, n4);
  cvt_wt_kernel<<<(EMB * EMB) / 256, 256, 0, stream>>>(Wq, Wqt);
  cvt_wt_kernel<<<(EMB * EMB) / 256, 256, 0, stream>>>(Wk, Wkt);
  cvt_wt_kernel<<<(EMB * EMB) / 256, 256, 0, stream>>>(Wv, Wvt);
  cvt_wt_kernel<<<(EMB * EMB) / 256, 256, 0, stream>>>(Wo, Wot);

  dim3 ggrid(BATCH * NSEQ / 128, EMB / 128);  // 32 x 8
  gemm_kernel<<<ggrid, 256, 0, stream>>>(q16, Wqt, bq, Qh, nullptr, 0);
  gemm_kernel<<<ggrid, 256, 0, stream>>>(k16, Wkt, bk, Kh, nullptr, 0);
  gemm_kernel<<<ggrid, 256, 0, stream>>>(v16, Wvt, bv, Vt, nullptr, 1);

  flash_attn_kernel<<<dim3(BATCH * HEADS, NSEQ / 128), 256, 0, stream>>>(
      Qh, Kh, Vt, Oh, mbuf, lbuf);

  gemm_kernel<<<ggrid, 256, 0, stream>>>(Oh, Wot, bo, nullptr, (float*)d_out, 2);

  attn_weights_kernel<<<dim3(NSEQ / 128, NSEQ / 128), 256, 0, stream>>>(
      Qh, Kh, mbuf, lbuf, (float*)d_out + (size_t)BATCH * NSEQ * EMB);
}
